// GroupedConformerBlock_5849745457793
// MI455X (gfx1250) — matrix operand order repaired and checked by reading, not compiled, not run on hardware
//
#include <hip/hip_runtime.h>
#include <stdint.h>

typedef __bf16 bf16_t;
typedef __attribute__((ext_vector_type(16))) __bf16 v16bf;
typedef __attribute__((ext_vector_type(8)))  float   v8f;

#ifndef __has_builtin
#define __has_builtin(x) 0
#endif
#if __has_builtin(__builtin_amdgcn_tensor_load_to_lds)
#define HAVE_TDM 1
#else
#define HAVE_TDM 0
#endif

#define DEV __device__ __forceinline__

DEV v8f wmma_bf16(v16bf a, v16bf b, v8f c) {
  return __builtin_amdgcn_wmma_f32_16x16x32_bf16(false, a, false, b, (short)0, c, false, false);
}

// A fragment: 16x32 (MxK) from row-major source, base at [0][k0], leading dim ld.
// lanes 0-15: row=lane, K 0..15 ; lanes 16-31: row=lane-16, K 16..31
DEV v16bf load_A_frag(const bf16_t* base, int ld, int lane) {
  int row = lane & 15;
  int koff = (lane < 16) ? 0 : 16;
  const bf16_t* p = base + row * ld + koff;
  v16bf a;
#pragma unroll
  for (int i = 0; i < 8; ++i) a[i] = p[i];
#pragma unroll
  for (int i = 0; i < 8; ++i) a[8 + i] = p[8 + i];
  return a;
}

// B fragment: 32x16 (KxN) from row-major KxN source, base at [k0][n0].
DEV v16bf load_B_frag(const bf16_t* base, int ldb, int lane) {
  int col = lane & 15;
  int krow = (lane < 16) ? 0 : 16;
  const bf16_t* p = base + krow * ldb + col;
  v16bf b;
#pragma unroll
  for (int i = 0; i < 16; ++i) b[i] = p[i * ldb];
  return b;
}

// B fragment where source is stored transposed (N x K): B[k][n] = src[n][k]; base at [n0][k0].
DEV v16bf load_B_fragT(const bf16_t* base, int ld, int lane) {
  int col = lane & 15;
  int krow = (lane < 16) ? 0 : 16;
  const bf16_t* p = base + col * ld + krow;
  v16bf b;
#pragma unroll
  for (int i = 0; i < 16; ++i) b[i] = p[i];
  return b;
}

// Stage a 32x16 B tile through per-wave LDS with transpose so the fragment read
// is contiguous: 2 coalesced 16B global loads per lane instead of 16 strided u16s.
DEV v16bf stage_B_frag(const bf16_t* g, int ldb, bf16_t* lbuf, int lane) {
  const uint4* src = (const uint4*)(g + (size_t)lane * ldb);   // row k = lane (32B)
  union { uint4 v4[2]; bf16_t e[16]; } u;
  u.v4[0] = src[0];
  u.v4[1] = src[1];
#pragma unroll
  for (int n = 0; n < 16; ++n) lbuf[n * 33 + lane] = u.e[n];   // store transposed [n][k]
  int col = lane & 15;
  int krow = (lane < 16) ? 0 : 16;
  const bf16_t* p = lbuf + col * 33 + krow;                    // contiguous 16 elems
  v16bf b;
#pragma unroll
  for (int i = 0; i < 16; ++i) b[i] = p[i];
  return b;
}

// ---------------------------------------------------------------------------
// TDM: one tensor_load_to_lds per attention K/V chunk.
// Tensor view: per (b): 8 "lines" (heads) of 8704 8-byte elements (1088 rows * 64 bf16),
// tile = 1024 x 8  (one 64-row x 64-col bf16 chunk per head), dest LDS contiguous.
#if HAVE_TDM
DEV void tdm_load_kv_chunk(unsigned lds_off, const bf16_t* gsrc) {
  typedef __attribute__((ext_vector_type(4))) unsigned u32x4;
  typedef __attribute__((ext_vector_type(8))) int i32x8;
  typedef __attribute__((ext_vector_type(4))) int i32x4;
  unsigned long long ga = (unsigned long long)(uintptr_t)gsrc;
  u32x4 g0;
  g0[0] = 1u;                                                  // count=1 (valid user D#)
  g0[1] = lds_off;                                             // LDS byte address
  g0[2] = (unsigned)ga;                                        // global_addr[31:0]
  g0[3] = (unsigned)((ga >> 32) & 0x01FFFFFFu) | (2u << 30);   // addr[56:32] | type=2
  const unsigned td0 = 8704u;   // tensor_dim0 / dim0_stride: 1088*64*2/8 (8B units)
  const unsigned tl0 = 1024u;   // tile_dim0: 64*64*2/8 (8B units)
  i32x8 g1;
  g1[0] = (int)(3u << 16);                                     // wg_mask=0, data_size=8B
  g1[1] = (int)((td0 & 0xFFFFu) << 16);                        // tensor_dim0 lo16
  g1[2] = (int)(((td0 >> 16) & 0xFFFFu) | (8u << 16));         // dim0 hi16 | tensor_dim1=8
  g1[3] = (int)((tl0 & 0xFFFFu) << 16);                        // dim1 hi16=0 | tile_dim0
  g1[4] = 8;                                                   // tile_dim1=8, tile_dim2=0
  g1[5] = (int)td0;                                            // tensor_dim0_stride lo32
  g1[6] = 0;                                                   // stride hi | dim1_stride lo
  g1[7] = 0;
  i32x4 z4 = {};
#if __clang_major__ >= 23
  i32x8 z8 = {};
  __builtin_amdgcn_tensor_load_to_lds(g0, g1, z4, z4, z8, 0);
#else
  __builtin_amdgcn_tensor_load_to_lds(g0, g1, z4, z4, 0);
#endif
}
#endif

// Issue one K/V chunk load (8 heads x 64 rows x 64 bf16 = 64KB) into LDS.
// src points at [b][h=0][j0][0] in the padded [b][8][1088][64] tensor.
// TDM path: async (wave 0 issues); fallback: synchronous vectorized copy.
DEV void kv_issue(bf16_t* dst, const bf16_t* src, int tid, int wv) {
#if HAVE_TDM
  if (wv == 0) tdm_load_kv_chunk((unsigned)(uintptr_t)dst, src);
  (void)tid;
#else
  uint4* d4 = (uint4*)dst;
  for (int t = tid; t < 4096; t += 256) {
    int h = t >> 9, r = t & 511;                 // 512 uint4 per head-chunk
    d4[t] = ((const uint4*)(src + (size_t)h * 1088 * 64))[r];
  }
  (void)wv;
#endif
}
DEV void kv_wait0(int wv) {
#if HAVE_TDM
  if (wv == 0) __builtin_amdgcn_s_wait_tensorcnt(0);
#endif
  (void)wv;
}
DEV void kv_wait1(int wv) {
#if HAVE_TDM
  if (wv == 0) __builtin_amdgcn_s_wait_tensorcnt(1);
#endif
  (void)wv;
}

// ---------------------------------------------------------------------------
// weight conversion
__global__ void cvt_kernel(const float* __restrict__ s, bf16_t* __restrict__ d, int n) {
  int i = blockIdx.x * 256 + threadIdx.x;
  if (i < n) d[i] = (bf16_t)s[i];
}
__global__ void cvt_t_kernel(const float* __restrict__ s, bf16_t* __restrict__ d, int R, int C) {
  int i = blockIdx.x * 256 + threadIdx.x;
  if (i < R * C) { int r = i / C, c = i % C; d[(size_t)c * R + r] = (bf16_t)s[i]; }
}
__global__ void mem_cvt_kernel(const float* __restrict__ mk, const float* __restrict__ mv,
                               bf16_t* __restrict__ ko, bf16_t* __restrict__ vo) {
  int t = threadIdx.x;              // 128 threads = 8 heads * 16 mem slots
  if (t >= 128) return;
  const float* kr = mk + t * 64;
  const float* vr = mv + t * 64;
  float ss = 0.f;
#pragma unroll
  for (int d = 0; d < 64; ++d) ss += kr[d] * kr[d];
  float inv = 1.f / fmaxf(sqrtf(ss), 1e-12f);
#pragma unroll
  for (int d = 0; d < 64; ++d) { ko[t * 64 + d] = (bf16_t)(kr[d] * inv); vo[t * 64 + d] = (bf16_t)vr[d]; }
}
// broadcast memory K/V into padded tensors and zero the tail rows (1040..1087)
__global__ void pad_fill_kernel(const bf16_t* __restrict__ kmem, const bf16_t* __restrict__ vmem,
                                bf16_t* __restrict__ Kpad, bf16_t* __restrict__ Vpad) {
  int i = blockIdx.x * 256 + threadIdx.x;
  if (i < 8 * 8 * 16 * 64) {
    int d = i & 63; int r = i >> 6; int m = r & 15; r >>= 4; int h = r & 7; int bb = r >> 3;
    size_t dst = (((size_t)bb * 8 + h) * 1088 + m) * 64 + d;
    Kpad[dst] = kmem[(h * 16 + m) * 64 + d];
    Vpad[dst] = vmem[(h * 16 + m) * 64 + d];
  }
  if (i < 8 * 8 * 48 * 64) {
    int d = i & 63; int r = i >> 6; int m = r % 48; r /= 48; int h = r & 7; int bb = r >> 3;
    size_t dst = (((size_t)bb * 8 + h) * 1088 + 1040 + m) * 64 + d;
    Kpad[dst] = (bf16_t)0.f;
    Vpad[dst] = (bf16_t)0.f;
  }
}

// ---------------------------------------------------------------------------
// layernorm: one block per row, write bf16 or f32
__global__ __launch_bounds__(256) void ln_kernel(const float* __restrict__ x,
                                                 const float* __restrict__ g,
                                                 const float* __restrict__ bp,
                                                 bf16_t* __restrict__ outb,
                                                 float* __restrict__ outf, int C) {
  __shared__ float red[256];
  __shared__ float sh[2];
  const int row = blockIdx.x, tid = threadIdx.x;
  const float* xr = x + (size_t)row * C;
  float s = 0.f;
  for (int c = tid; c < C; c += 256) s += xr[c];
  red[tid] = s; __syncthreads();
  for (int st = 128; st > 0; st >>= 1) { if (tid < st) red[tid] += red[tid + st]; __syncthreads(); }
  if (tid == 0) sh[0] = red[0] / (float)C;
  __syncthreads();
  float m = sh[0];
  float v = 0.f;
  for (int c = tid; c < C; c += 256) { float d = xr[c] - m; v += d * d; }
  red[tid] = v; __syncthreads();
  for (int st = 128; st > 0; st >>= 1) { if (tid < st) red[tid] += red[tid + st]; __syncthreads(); }
  if (tid == 0) sh[1] = rsqrtf(red[0] / (float)C + 1e-5f);
  __syncthreads();
  float rs = sh[1];
  for (int c = tid; c < C; c += 256) {
    float y = (xr[c] - m) * rs * g[c] + bp[c];
    if (outb) outb[(size_t)row * C + c] = (bf16_t)y;
    else      outf[(size_t)row * C + c] = y;
  }
}

// ---------------------------------------------------------------------------
// generic WMMA GEMM: block = 16 rows x 128 cols (8 waves, 16x16 tile each)
// op: 0 -> store f32 ; 1 -> store bf16 ; 2 -> outF = res + acc + bias
__global__ __launch_bounds__(256) void gemm_kernel(const bf16_t* __restrict__ A,
                                                   const bf16_t* __restrict__ B,
                                                   const float* __restrict__ bias,
                                                   const float* __restrict__ res,
                                                   float* __restrict__ outF,
                                                   bf16_t* __restrict__ outB,
                                                   int M, int N, int K, int op) {
  __shared__ unsigned aTileU[256];          // 16x32 bf16
  __shared__ bf16_t bTile[8][16 * 33];      // per-wave transposed B tile
  const int tid = threadIdx.x, lane = tid & 31, wv = tid >> 5;
  const int m0 = blockIdx.y * 16;
  const int n0 = blockIdx.x * 128 + wv * 16;
  (void)M;
  v8f acc = {};
  for (int k0 = 0; k0 < K; k0 += 32) {
    __syncthreads();
    {
      int r = tid >> 4, c2 = tid & 15;      // one b32 per thread
      aTileU[tid] = ((const unsigned*)(A + (size_t)(m0 + r) * K + k0))[c2];
    }
    __syncthreads();
    if (k0 + 32 < K) __builtin_prefetch(B + (size_t)(k0 + 32) * N + n0, 0, 1);
    v16bf af = load_A_frag((const bf16_t*)aTileU, 32, lane);
    v16bf bf = stage_B_frag(B + (size_t)k0 * N + n0, N, bTile[wv], lane);
    acc = wmma_bf16(af, bf, acc);
  }
  const int col = n0 + (lane & 15);
  const float bv = bias ? bias[col] : 0.f;
  const int rb = (lane < 16) ? 0 : 8;
#pragma unroll
  for (int r = 0; r < 8; ++r) {
    size_t idx = (size_t)(m0 + rb + r) * N + col;
    float v = acc[r] + bv;
    if (op == 0)      outF[idx] = v;
    else if (op == 1) outB[idx] = (bf16_t)v;
    else              outF[idx] = res[idx] + v;
  }
}

// ---------------------------------------------------------------------------
// fused LN -> GEMM(512->2048)+swish (hidden in LDS, bf16) -> GEMM(2048->512) -> 0.5*acc + x
__global__ __launch_bounds__(256) void ffn_kernel(float* __restrict__ X,
                                                  const float* __restrict__ lng,
                                                  const float* __restrict__ lnb,
                                                  const bf16_t* __restrict__ W1,
                                                  const float* __restrict__ b1,
                                                  const bf16_t* __restrict__ W2,
                                                  const float* __restrict__ b2) {
  extern __shared__ char smem_raw[];
  bf16_t* xln = (bf16_t*)smem_raw;          // 16*512
  bf16_t* hid = xln + 16 * 512;             // 16*2048
  bf16_t* btile = hid + 16 * 2048;          // 8 waves * 16*33
  float* red = (float*)(btile + 8 * 16 * 33 + 8);  // 256 (+8 bf16 pad for 4B align)
  float* muv = red + 256;                   // 16
  float* rsv = muv + 16;                    // 16
  const int tid = threadIdx.x, lane = tid & 31, wv = tid >> 5;
  const int m0 = blockIdx.x * 16;
  const int r16 = tid >> 4, s16 = tid & 15;
  bf16_t* mybt = btile + wv * 16 * 33;
  const float* xr = X + (size_t)(m0 + r16) * 512;

  float s = 0.f;
  for (int c = s16; c < 512; c += 16) s += xr[c];
  red[tid] = s; __syncthreads();
  if (s16 == 0) {
    float tot = 0.f;
    for (int k = 0; k < 16; ++k) tot += red[r16 * 16 + k];
    muv[r16] = tot * (1.f / 512.f);
  }
  __syncthreads();
  float m = muv[r16];
  float v = 0.f;
  for (int c = s16; c < 512; c += 16) { float d = xr[c] - m; v += d * d; }
  red[tid] = v; __syncthreads();
  if (s16 == 0) {
    float tot = 0.f;
    for (int k = 0; k < 16; ++k) tot += red[r16 * 16 + k];
    rsv[r16] = rsqrtf(tot * (1.f / 512.f) + 1e-5f);
  }
  __syncthreads();
  float rs = rsv[r16];
  for (int c = s16; c < 512; c += 16)
    xln[r16 * 512 + c] = (bf16_t)((xr[c] - m) * rs * lng[c] + lnb[c]);
  __syncthreads();

  // GEMM1 + swish -> hid
  for (int t = 0; t < 16; ++t) {
    int n0 = wv * 256 + t * 16;
    v8f acc = {};
    for (int k0 = 0; k0 < 512; k0 += 32) {
      v16bf a = load_A_frag(xln + k0, 512, lane);
      v16bf bf = stage_B_frag(W1 + (size_t)k0 * 2048 + n0, 2048, mybt, lane);
      acc = wmma_bf16(a, bf, acc);
    }
    int col = n0 + (lane & 15);
    float bv = b1[col];
    int rb = (lane < 16) ? 0 : 8;
#pragma unroll
    for (int r = 0; r < 8; ++r) {
      float h = acc[r] + bv;
      h = h / (1.f + __expf(-h));
      hid[(rb + r) * 2048 + col] = (bf16_t)h;
    }
  }
  __syncthreads();

  // GEMM2 + residual
  for (int t = 0; t < 4; ++t) {
    int n0 = wv * 64 + t * 16;
    v8f acc = {};
    for (int k0 = 0; k0 < 2048; k0 += 32) {
      v16bf a = load_A_frag(hid + k0, 2048, lane);
      v16bf bf = stage_B_frag(W2 + (size_t)k0 * 512 + n0, 512, mybt, lane);
      acc = wmma_bf16(a, bf, acc);
    }
    int col = n0 + (lane & 15);
    float bv = b2[col];
    int rb = (lane < 16) ? 0 : 8;
#pragma unroll
    for (int r = 0; r < 8; ++r) {
      size_t idx = (size_t)(m0 + rb + r) * 512 + col;
      X[idx] = X[idx] + 0.5f * (acc[r] + bv);
    }
  }
}

// ---------------------------------------------------------------------------
// per-head l2norm (optional) + scale + transpose to [b][h][rowstride rows][64] bf16
__global__ __launch_bounds__(256) void qkv_cvt_kernel(const float* __restrict__ src,
                                                      bf16_t* __restrict__ dst,
                                                      int donorm, float scale,
                                                      int rowstride, int rowoff) {
  int t = blockIdx.x * 256 + threadIdx.x;
  if (t >= 8 * 1024 * 8) return;
  int h = t & 7;
  int row = t >> 3;              // b*1024 + i
  int bb = row >> 10;
  int ii = row & 1023;
  const float* p = src + (size_t)row * 512 + h * 64;
  float inv = scale;
  if (donorm) {
    float ss = 0.f;
#pragma unroll
    for (int d = 0; d < 64; ++d) ss += p[d] * p[d];
    inv = scale / fmaxf(sqrtf(ss), 1e-12f);
  }
  bf16_t* q = dst + (((size_t)bb * 8 + h) * rowstride + rowoff + ii) * 64;
#pragma unroll
  for (int d = 0; d < 64; ++d) q[d] = (bf16_t)(p[d] * inv);
}

// ---------------------------------------------------------------------------
// fused talking-heads attention with memory KV, approximate top-64 threshold,
// and double-buffered TDM chunk staging (compute overlaps tensor DMA).
__global__ __launch_bounds__(256) void attn_kernel(const bf16_t* __restrict__ qn,
                                                   const bf16_t* __restrict__ Kpad,
                                                   const bf16_t* __restrict__ Vpad,
                                                   const float* __restrict__ pre,
                                                   const float* __restrict__ post,
                                                   const int* __restrict__ lengths,
                                                   bf16_t* __restrict__ O) {
  constexpr int NG = 1024;
  constexpr int NJ = 1040;             // 16 memory + 1024 keys
  constexpr int NCH = 17;              // 17 chunks of 64 (padded to 1088)
  constexpr float NEGF = -1.0e9f;
  constexpr float RANGE = 12.8f;
  constexpr int NB = 32;
  constexpr float BINW = RANGE / (float)NB;

  extern __shared__ char smem_raw[];
  bf16_t* qs  = (bf16_t*)smem_raw;              // [8][16][64]
  bf16_t* kvb0 = qs + 8 * 16 * 64;              // [8][64][64]  ping
  bf16_t* kvb1 = kvb0 + 8 * 64 * 64;            // [8][64][64]  pong
  bf16_t* at2 = kvb1 + 8 * 64 * 64;             // [8][16][64]
  float* dots = (float*)(at2 + 8 * 16 * 64);    // [8][16][64] raw per-head dots
  float* rowmax = dots + 8 * 16 * 64;           // [8][16]
  float* zsum = rowmax + 128;                   // [8][16]
  float* thr_s = zsum + 128;                    // [8][16]
  float* preS = thr_s + 128;                    // [8][8]
  float* postS = preS + 64;                     // [8][8]
  unsigned* hist = (unsigned*)(postS + 64);     // [8][16][NB]

  const int tid = threadIdx.x, lane = tid & 31, wv = tid >> 5;
  const int b = blockIdx.y;
  const int i0 = blockIdx.x * 16;
  const int nlen = (lengths[b] + 1) >> 1;       // ceil(len / 2)

  if (tid < 64) { preS[tid] = pre[tid]; postS[tid] = post[tid]; }
  if (tid < 128) rowmax[tid] = -3.0e38f;
  {
    uint4* qd = (uint4*)qs;
    for (int t = tid; t < 1024; t += 256) {      // 16KB in b128 chunks
      int d4i = t & 7, r = t >> 3, i = r & 15, h = r >> 4;
      qd[t] = ((const uint4*)(qn + (((size_t)b * 8 + h) * 1024 + i0 + i) * 64))[d4i];
    }
  }
  __syncthreads();

  const int ii = lane & 15;
  const int jhh = lane >> 4;
  const bool qok = (i0 + ii) < nlen;
  v8f oacc[4] = {};
  const bf16_t* Kb = Kpad + (size_t)b * 8 * 1088 * 64;
  const bf16_t* Vb = Vpad + (size_t)b * 8 * 1088 * 64;

  // wave wv computes raw per-head dot tiles for head wv from the given K buffer
  auto compute_dots = [&](const bf16_t* kall) {
    const bf16_t* qbase = qs + wv * 16 * 64;
    const bf16_t* kbase = kall + wv * 64 * 64;
#pragma unroll
    for (int jt = 0; jt < 4; ++jt) {
      v8f acc = {};
#pragma unroll
      for (int k0 = 0; k0 < 64; k0 += 32) {
        v16bf a = load_A_frag(qbase + k0, 64, lane);
        v16bf bm = load_B_fragT(kbase + (jt * 16) * 64 + k0, 64, lane);
        acc = wmma_bf16(a, bm, acc);
      }
      int col = lane & 15;
      int rb = (lane < 16) ? 0 : 8;
#pragma unroll
      for (int r = 0; r < 8; ++r)
        dots[(wv * 16 + rb + r) * 64 + jt * 16 + col] = acc[r];
    }
  };

  for (int pass = 0; pass < 4; ++pass) {
    if (pass == 1) {
      for (int t = tid; t < 8 * 16 * NB; t += 256) hist[t] = 0u;
    } else if (pass == 2) {
      if (tid < 128) {
        float m = rowmax[tid];
        float lo = m - RANGE;
        unsigned cnt = 0;
        float tv = -3.0e38f;
        for (int bin = NB - 1; bin >= 0; --bin) {
          cnt += hist[tid * NB + bin];
          if (cnt >= 64u) { tv = lo + bin * BINW; break; }
        }
        thr_s[tid] = tv;
        zsum[tid] = 0.f;
      }
    }
    __syncthreads();

    if (pass < 3) {
      // ---- passes 0..2: K only, ping-pong double buffer ----
      kv_issue(kvb0, Kb, tid, wv);                     // K(0)
      for (int ch = 0; ch < NCH; ++ch) {
        const int j0 = ch * 64;
        bf16_t* cur = (ch & 1) ? kvb1 : kvb0;
        bf16_t* nxt = (ch & 1) ? kvb0 : kvb1;
        if (ch + 1 < NCH) { kv_issue(nxt, Kb + (size_t)(j0 + 64) * 64, tid, wv); kv_wait1(wv); }
        else              { kv_wait0(wv); }
        __syncthreads();
        compute_dots(cur);
        __syncthreads();

        const int g = wv;
        if (pass == 0) {
          float lmax = -3.0e38f;
          for (int jj = 0; jj < 32; ++jj) {
            int j = jhh * 32 + jj, jg = j0 + j;
            if (jg >= NJ) continue;
            float dm = 0.f;
#pragma unroll
            for (int h = 0; h < 8; ++h) dm += preS[h * 8 + g] * dots[(h * 16 + ii) * 64 + j];
            bool kok = (jg < 16) || ((jg - 16) < nlen);
            if (!(qok && kok)) dm = NEGF;
            lmax = fmaxf(lmax, dm);
          }
          lmax = fmaxf(lmax, __shfl_xor(lmax, 16, 32));
          if (lane < 16) rowmax[g * 16 + ii] = fmaxf(rowmax[g * 16 + ii], lmax);
        } else if (pass == 1) {
          float m = rowmax[g * 16 + ii];
          float lo = m - RANGE;
          for (int jj = 0; jj < 32; ++jj) {
            int j = jhh * 32 + jj, jg = j0 + j;
            if (jg >= NJ) continue;
            float dm = 0.f;
#pragma unroll
            for (int h = 0; h < 8; ++h) dm += preS[h * 8 + g] * dots[(h * 16 + ii) * 64 + j];
            bool kok = (jg < 16) || ((jg - 16) < nlen);
            if (!(qok && kok)) dm = NEGF;
            int bin = (int)((dm - lo) * (1.f / BINW));
            bin = bin < 0 ? 0 : (bin > NB - 1 ? NB - 1 : bin);
            atomicAdd(&hist[(g * 16 + ii) * NB + bin], 1u);
          }
        } else {
          float m = rowmax[g * 16 + ii];
          float tv = thr_s[g * 16 + ii];
          float z = 0.f;
          for (int jj = 0; jj < 32; ++jj) {
            int j = jhh * 32 + jj, jg = j0 + j;
            if (jg >= NJ) continue;
            float dm = 0.f;
#pragma unroll
            for (int h = 0; h < 8; ++h) dm += preS[h * 8 + g] * dots[(h * 16 + ii) * 64 + j];
            bool kok = (jg < 16) || ((jg - 16) < nlen);
            if (!(qok && kok)) dm = NEGF;
            if (dm >= tv) z += __expf(dm - m);
          }
          z += __shfl_xor(z, 16, 32);
          if (lane < 16) zsum[g * 16 + ii] += z;
        }
        __syncthreads();
      }
    } else {
      // ---- pass 3: K (buf0) + V (buf1) software pipeline ----
      kv_issue(kvb0, Kb, tid, wv);                     // K(0)
      for (int ch = 0; ch < NCH; ++ch) {
        const int j0 = ch * 64;
        kv_issue(kvb1, Vb + (size_t)j0 * 64, tid, wv); // V(ch)
        kv_wait1(wv);                                  // K(ch) landed
        __syncthreads();
        compute_dots(kvb0);
        __syncthreads();

        const int g = wv;
        {
          float mh[8], th[8], zi[8], po[8];
#pragma unroll
          for (int h = 0; h < 8; ++h) {
            mh[h] = rowmax[h * 16 + ii];
            th[h] = thr_s[h * 16 + ii];
            zi[h] = 1.f / fmaxf(zsum[h * 16 + ii], 1e-30f);
            po[h] = postS[h * 8 + g];
          }
          for (int jj = 0; jj < 32; ++jj) {
            int j = jhh * 32 + jj, jg = j0 + j;
            float a2 = 0.f;
            if (jg < NJ) {
              float raw[8];
#pragma unroll
              for (int h0 = 0; h0 < 8; ++h0) raw[h0] = dots[(h0 * 16 + ii) * 64 + j];
              bool kok = (jg < 16) || ((jg - 16) < nlen);
              bool valid = qok && kok;
#pragma unroll
              for (int h = 0; h < 8; ++h) {
                float dm = 0.f;
#pragma unroll
                for (int h0 = 0; h0 < 8; ++h0) dm += preS[h0 * 8 + h] * raw[h0];
                if (!valid) dm = NEGF;
                float e = (dm >= th[h]) ? __expf(dm - mh[h]) * zi[h] : 0.f;
                a2 += po[h] * e;
              }
            }
            at2[(g * 16 + ii) * 64 + j] = (bf16_t)a2;
          }
        }
        __syncthreads();                               // dots-phase reads of kvb0 done

        if (ch + 1 < NCH) { kv_issue(kvb0, Kb + (size_t)(j0 + 64) * 64, tid, wv); kv_wait1(wv); }
        else              { kv_wait0(wv); }            // V(ch) landed
        __syncthreads();

        const bf16_t* abase = at2 + wv * 16 * 64;
        const bf16_t* vbase = kvb1 + wv * 64 * 64;
#pragma unroll
        for (int dt = 0; dt < 4; ++dt) {
#pragma unroll
          for (int k0 = 0; k0 < 64; k0 += 32) {
            v16bf a = load_A_frag(abase + k0, 64, lane);
            v16bf bm = load_B_frag(vbase + k0 * 64 + dt * 16, 64, lane);
            oacc[dt] = wmma_bf16(a, bm, oacc[dt]);
          }
        }
        __syncthreads();                               // AV reads of kvb1 done
      }
    }
  }

  // write O[b][i][h*64+d] bf16
  {
    int col = lane & 15;
    int rb = (lane < 16) ? 0 : 8;
#pragma unroll
    for (int dt = 0; dt < 4; ++dt)
#pragma unroll
      for (int r = 0; r < 8; ++r)
        O[((size_t)b * NG + i0 + rb + r) * 512 + wv * 64 + dt * 16 + col] = (bf16_t)oacc[dt][r];
  }
}

// ---------------------------------------------------------------------------
// GLU + transpose to [b][c][n] fp32
__global__ void glu_kernel(const float* __restrict__ t, float* __restrict__ z) {
  size_t i = (size_t)blockIdx.x * 256 + threadIdx.x;
  if (i >= (size_t)16384 * 512) return;
  int c = (int)(i & 511);
  size_t row = i >> 9;            // b*2048 + n
  size_t bb = row >> 11;
  size_t nn = row & 2047;
  float a = t[row * 1024 + c];
  float gg = t[row * 1024 + 512 + c];
  z[((bb * 512 + c) * 2048) + nn] = a * (1.f / (1.f + __expf(-gg)));
}

// depthwise conv (K=31, SAME) + bias + bn-scale + swish, write [b][n][c] bf16
__global__ __launch_bounds__(256) void dwconv_kernel(const float* __restrict__ z,
                                                     const float* __restrict__ dw,
                                                     const float* __restrict__ dwb,
                                                     const float* __restrict__ bng,
                                                     const float* __restrict__ bnb,
                                                     bf16_t* __restrict__ out) {
  __shared__ float rowb[2048 + 30];
  __shared__ float taps[31];
  const int bc = blockIdx.x;
  const int b = bc >> 9, c = bc & 511;
  const int tid = threadIdx.x;
  const float* zr = z + ((size_t)b * 512 + c) * 2048;
  for (int i = tid; i < 2078; i += 256) {
    int n = i - 15;
    rowb[i] = (n >= 0 && n < 2048) ? zr[n] : 0.f;
  }
  if (tid < 31) taps[tid] = dw[c * 31 + tid];
  __syncthreads();
  const float bias = dwb[c];
  const float scale = rsqrtf(1.0f + 1e-5f) * bng[c];
  const float shift = bnb[c];
  for (int n = tid; n < 2048; n += 256) {
    float acc = 0.f;
#pragma unroll
    for (int t = 0; t < 31; ++t) acc += rowb[n + t] * taps[t];
    acc += bias;
    acc = acc * scale + shift;
    acc = acc / (1.f + __expf(-acc));
    out[((size_t)b * 2048 + n) * 512 + c] = (bf16_t)acc;
  }
}

// ---------------------------------------------------------------------------
extern "C" void kernel_launch(void* const* d_in, const int* in_sizes, int n_in,
                              void* d_out, int out_size, void* d_ws, size_t ws_size,
                              hipStream_t stream) {
  (void)in_sizes; (void)n_in; (void)out_size; (void)ws_size;
  const float* x       = (const float*)d_in[0];
  const int*   lengths = (const int*)d_in[1];
  const float* ln1_g = (const float*)d_in[2];
  const float* ln1_b = (const float*)d_in[3];
  const float* ff1_w1 = (const float*)d_in[4];
  const float* ff1_b1 = (const float*)d_in[5];
  const float* ff1_w2 = (const float*)d_in[6];
  const float* ff1_b2 = (const float*)d_in[7];
  const float* aln_g = (const float*)d_in[8];
  const float* aln_b = (const float*)d_in[9];
  const float* Wq = (const float*)d_in[10];
  const float* Wk = (const float*)d_in[11];
  const float* Wv = (const float*)d_in[12];
  const float* mem_k = (const float*)d_in[13];
  const float* mem_v = (const float*)d_in[14];
  const float* pre_t = (const float*)d_in[15];
  const float* post_t = (const float*)d_in[16];
  const float* Wo = (const float*)d_in[17];
  const float* bo = (const float*)d_in[18];
  const float* cln_g = (const float*)d_in[19];
  const float* cln_b = (const float*)d_in[20];
  const float* cw1 = (const float*)d_in[21];
  const float* cb1 = (const float*)d_in[22];
  const float* dwW = (const float*)d_in[23];
  const float* dwb = (const float*)d_in[24];
  const float* bn_g = (const float*)d_in[25];
  const float* bn_b = (const float*)d_in[26];
  const float* cw2 = (const float*)d_in[27];
  const float* cb2 = (const float*)d_in[28];
  const float* ln2_g = (const float*)d_in[29];
  const float* ln2_b = (const float*)d_in[30];
  const float* ff2_w1 = (const float*)d_in[31];
  const float* ff2_b1 = (const float*)d_in[32];
  const float* ff2_w2 = (const float*)d_in[33];
  const float* ff2_b2 = (const float*)d_in[34];
  const float* pn_g = (const float*)d_in[35];
  const float* pn_b = (const float*)d_in[36];

  const int NT = 16384;  // 8 * 2048 rows
  char* wsp = (char*)d_ws;
  size_t off = 0;
  auto alloc = [&](size_t bytes) -> void* {
    void* p = (void*)(wsp + off);
    off += (bytes + 255) & ~(size_t)255;
    return p;
  };

  float*  X    = (float*) alloc((size_t)NT * 512 * 4);
  float*  tmpF = (float*) alloc((size_t)NT * 1024 * 4);
  bf16_t* Ybf  = (bf16_t*)alloc((size_t)NT * 512 * 2);       // also viewed as 8192x1024
  float*  Z    = (float*) alloc((size_t)8 * 512 * 2048 * 4);
  bf16_t* Dw   = (bf16_t*)alloc((size_t)NT * 512 * 2);
  bf16_t* Obuf = (bf16_t*)alloc((size_t)8192 * 512 * 2);
  bf16_t* Qb   = (bf16_t*)alloc((size_t)8 * 8 * 1024 * 64 * 2);
  bf16_t* Kpad = (bf16_t*)alloc((size_t)8 * 8 * 1088 * 64 * 2);
  bf16_t* Vpad = (bf16_t*)alloc((size_t)8 * 8 * 1088 * 64 * 2);
  bf16_t* Kmem = (bf16_t*)alloc((size_t)8 * 16 * 64 * 2);
  bf16_t* Vmem = (bf16_t*)alloc((size_t)8 * 16 * 64 * 2);
  bf16_t* W1a  = (bf16_t*)alloc((size_t)512 * 2048 * 2);
  bf16_t* W2a  = (bf16_t*)alloc((size_t)2048 * 512 * 2);
  bf16_t* W1b  = (bf16_t*)alloc((size_t)512 * 2048 * 2);
  bf16_t* W2b  = (bf16_t*)alloc((size_t)2048 * 512 * 2);
  bf16_t* Wqb  = (bf16_t*)alloc((size_t)1024 * 512 * 2);
  bf16_t* Wkb  = (bf16_t*)alloc((size_t)1024 * 512 * 2);
  bf16_t* Wvb  = (bf16_t*)alloc((size_t)1024 * 512 * 2);
  bf16_t* Wob  = (bf16_t*)alloc((size_t)512 * 1024 * 2);
  bf16_t* Cw1b = (bf16_t*)alloc((size_t)512 * 1024 * 2);     // transposed cw1
  bf16_t* Cw2b = (bf16_t*)alloc((size_t)512 * 512 * 2);      // transposed cw2

  const size_t FFN_SMEM = (size_t)(16 * 512 + 16 * 2048 + 8 * 16 * 33 + 8) * sizeof(bf16_t)
                        + (size_t)(256 + 16 + 16) * sizeof(float);
  const size_t ATTN_SMEM = (size_t)(8 * 16 * 64 + 2 * 8 * 64 * 64 + 8 * 16 * 64) * sizeof(bf16_t)
                         + (size_t)(8 * 16 * 64 + 128 * 3 + 64 * 2) * sizeof(float)
                         + (size_t)(8 * 16 * 32) * sizeof(unsigned);

  // running activation buffer
  hipMemcpyAsync(X, x, (size_t)NT * 512 * 4, hipMemcpyDeviceToDevice, stream);

  // weight conversions
  auto cvt = [&](const float* s, bf16_t* d, int n) {
    cvt_kernel<<<(n + 255) / 256, 256, 0, stream>>>(s, d, n);
  };
  cvt(ff1_w1, W1a, 512 * 2048);  cvt(ff1_w2, W2a, 2048 * 512);
  cvt(ff2_w1, W1b, 512 * 2048);  cvt(ff2_w2, W2b, 2048 * 512);
  cvt(Wq, Wqb, 1024 * 512);      cvt(Wk, Wkb, 1024 * 512);
  cvt(Wv, Wvb, 1024 * 512);      cvt(Wo, Wob, 512 * 1024);
  cvt_t_kernel<<<(1024 * 512 + 255) / 256, 256, 0, stream>>>(cw1, Cw1b, 1024, 512);
  cvt_t_kernel<<<(512 * 512 + 255) / 256, 256, 0, stream>>>(cw2, Cw2b, 512, 512);
  mem_cvt_kernel<<<1, 128, 0, stream>>>(mem_k, mem_v, Kmem, Vmem);

  // FFN1 (macaron half)
  ffn_kernel<<<NT / 16, 256, FFN_SMEM, stream>>>(X, ln1_g, ln1_b, W1a, ff1_b1, W2a, ff1_b2);

  // attention: grouped LN (DS=1024, contiguous reshape) -> QKV -> fused attention -> Wo + residual
  ln_kernel<<<8192, 256, 0, stream>>>(X, aln_g, aln_b, Ybf, nullptr, 1024);
  gemm_kernel<<<dim3(4, 512), 256, 0, stream>>>(Ybf, Wqb, nullptr, nullptr, tmpF, nullptr, 8192, 512, 1024, 0);
  qkv_cvt_kernel<<<256, 256, 0, stream>>>(tmpF, Qb, 1, 10.0f, 1024, 0);     // fold QK_SCALE into q
  gemm_kernel<<<dim3(4, 512), 256, 0, stream>>>(Ybf, Wkb, nullptr, nullptr, tmpF, nullptr, 8192, 512, 1024, 0);
  qkv_cvt_kernel<<<256, 256, 0, stream>>>(tmpF, Kpad, 1, 1.0f, 1088, 16);
  gemm_kernel<<<dim3(4, 512), 256, 0, stream>>>(Ybf, Wvb, nullptr, nullptr, tmpF, nullptr, 8192, 512, 1024, 0);
  qkv_cvt_kernel<<<256, 256, 0, stream>>>(tmpF, Vpad, 0, 1.0f, 1088, 16);
  pad_fill_kernel<<<768, 256, 0, stream>>>(Kmem, Vmem, Kpad, Vpad);
  attn_kernel<<<dim3(64, 8), 256, ATTN_SMEM, stream>>>(Qb, Kpad, Vpad, pre_t, post_t, lengths, Obuf);
  gemm_kernel<<<dim3(8, 512), 256, 0, stream>>>(Obuf, Wob, bo, X, X, nullptr, 8192, 1024, 512, 2);

  // conv module
  ln_kernel<<<NT, 256, 0, stream>>>(X, cln_g, cln_b, Ybf, nullptr, 512);
  gemm_kernel<<<dim3(8, NT / 16), 256, 0, stream>>>(Ybf, Cw1b, cb1, nullptr, tmpF, nullptr, NT, 1024, 512, 0);
  glu_kernel<<<(NT * 512) / 256, 256, 0, stream>>>(tmpF, Z);
  dwconv_kernel<<<8 * 512, 256, 0, stream>>>(Z, dwW, dwb, bn_g, bn_b, Dw);
  gemm_kernel<<<dim3(4, NT / 16), 256, 0, stream>>>(Dw, Cw2b, cb2, X, X, nullptr, NT, 512, 512, 2);

  // FFN2 + final LN
  ffn_kernel<<<NT / 16, 256, FFN_SMEM, stream>>>(X, ln2_g, ln2_b, W1b, ff2_b1, W2b, ff2_b2);
  ln_kernel<<<NT, 256, 0, stream>>>(X, pn_g, pn_b, nullptr, (float*)d_out, 512);
}
